// WindowAttention_14422500180622
// MI455X (gfx1250) — matrix operand order repaired and checked by reading, not compiled, not run on hardware
//
#include <hip/hip_runtime.h>
#include <hip/hip_bf16.h>

typedef __attribute__((ext_vector_type(4)))  float  v4f;
typedef __attribute__((ext_vector_type(8)))  float  v8f;
typedef __attribute__((ext_vector_type(4)))  __bf16 v4bf;
typedef __attribute__((ext_vector_type(8)))  __bf16 v8bf;
typedef __attribute__((ext_vector_type(16))) __bf16 v16bf;

#define NTOK   49
#define CIN    1024
#define DIMV   512
#define NHEAD  8
#define HDIM   64
#define NWIN   256
#define SCALEF 0.125f

// ---- workspace layout (in bf16 elements) ----
#define WQT_OFF 0
#define WKT_OFF (512*1024)
#define WVT_OFF (2*512*1024)
#define WPT_OFF (2*512*1024 + 512*512)
#define WS_HALVES (WPT_OFF + 1024*512)

// ---- LDS layout (bytes) ----
#define XS     1032                 // xb row stride in halves (padded vs 1024)
#define XB_OFF 0                    // 50 rows * 1032 * 2 = 103200
#define QS     72                   // Q/K/Vt/P row stride in halves
#define QH_OFF 103200               // 64*72*2 = 9216
#define KH_OFF 112416
#define VT_OFF 121632
#define SS     68                   // score row stride in floats
#define S_OFF  130848               // 64*68*4 = 17408
#define PH_OFF 148256               // 9216
#define OS     520                  // attn-out row stride in halves
#define OB_OFF 157472               // 64*520*2 = 66560
#define MKS    52                   // mask row stride in floats
#define MK_OFF 224032               // 49*52*4 = 10192
#define LDS_BYTES 234224

// A-operand (16x32 bf16): lane m = lane&15, K chunks of 8 at (lane>>4)*8 and +16
static __device__ __forceinline__ v16bf load_a16(const __bf16* p) {
  union { v16bf w; v8bf h[2]; } u;
  u.h[0] = *(const v8bf*)(p);
  u.h[1] = *(const v8bf*)(p + 16);
  return u.w;
}
// B-operand (32x16 bf16): lane n = lane&15, K chunks of 8 at (lane>>4)*8 and +16
static __device__ __forceinline__ v16bf load_b16(const __bf16* p) {
  union { v16bf w; v8bf h[2]; } u;
  u.h[0] = *(const v8bf*)(p);
  u.h[1] = *(const v8bf*)(p + 16);
  return u.w;
}
static __device__ __forceinline__ v8f wmma_bf16(v16bf a, v16bf b, v8f c) {
  return __builtin_amdgcn_wmma_f32_16x16x32_bf16(false, a, false, b, (short)0, c,
                                                 false, false);
}

// ---- prep: transpose + convert all weights to bf16, W^T[n][k] ----
__global__ __launch_bounds__(256)
void wattn_prep(const float* __restrict__ Wq, const float* __restrict__ Wk,
                const float* __restrict__ Wv, const float* __restrict__ Wp,
                __bf16* __restrict__ ws) {
  int idx = blockIdx.x * 256 + threadIdx.x;
  if (idx < WKT_OFF) {                       // WqT: 512 x 1024
    int n = idx >> 10, k = idx & 1023;
    ws[idx] = (__bf16)Wq[k * 512 + n];
  } else if (idx < WVT_OFF) {                // WkT: 512 x 1024
    int r = idx - WKT_OFF;
    int n = r >> 10, k = r & 1023;
    ws[idx] = (__bf16)Wk[k * 512 + n];
  } else if (idx < WPT_OFF) {                // WvT: 512 x 512
    int r = idx - WVT_OFF;
    int n = r >> 9, k = r & 511;
    ws[idx] = (__bf16)Wv[k * 512 + n];
  } else if (idx < WS_HALVES) {              // WpT: 1024 x 512
    int r = idx - WPT_OFF;
    int n = r >> 9, k = r & 511;
    ws[idx] = (__bf16)Wp[k * 1024 + n];
  }
}

// ---- fused window attention: one block per window ----
__global__ __launch_bounds__(384)
void wattn_main(const float* __restrict__ x, const float* __restrict__ mask,
                const float* __restrict__ bq, const float* __restrict__ bk,
                const float* __restrict__ bv, const float* __restrict__ bp,
                const __bf16* __restrict__ wst, float* __restrict__ out) {
  extern __shared__ __align__(16) unsigned char smem[];
  __bf16* xb = (__bf16*)(smem + XB_OFF);
  __bf16* qh = (__bf16*)(smem + QH_OFF);
  __bf16* kh = (__bf16*)(smem + KH_OFF);
  __bf16* vt = (__bf16*)(smem + VT_OFF);
  float*  sb = (float*)(smem + S_OFF);
  __bf16* ph = (__bf16*)(smem + PH_OFF);
  __bf16* ob = (__bf16*)(smem + OB_OFF);
  float*  mk = (float*)(smem + MK_OFF);

  const int tid  = threadIdx.x;
  const int lane = tid & 31;
  const int wv   = tid >> 5;     // 0..11
  const int b    = blockIdx.x;
  const int win  = b & (NWIN - 1);
  const int l15  = lane & 15;
  const int hi   = lane >> 4;
  const int koa  = hi * 8;       // A-operand K base
  const int kob  = hi * 8;       // B-operand K base

  // ---- stage x[b] -> bf16 LDS (packed b64 stores), zero row 49, mask window ----
  {
    const v4f* xsrc = (const v4f*)(x + (size_t)b * NTOK * CIN);
    for (int idx = tid; idx < NTOK * (CIN / 4); idx += 384) {
      int row = idx >> 8;        // CIN/4 == 256
      int c4  = idx & 255;
      v4f v = xsrc[row * 256 + c4];
      v4bf o;
      o[0] = (__bf16)v[0]; o[1] = (__bf16)v[1];
      o[2] = (__bf16)v[2]; o[3] = (__bf16)v[3];
      *(v4bf*)(xb + row * XS + c4 * 4) = o;
    }
    v4bf z = {};
    for (int c4 = tid; c4 < XS / 4; c4 += 384)
      *(v4bf*)(xb + 49 * XS + c4 * 4) = z;
    const float* msrc = mask + (size_t)win * NTOK * NTOK;
    for (int idx = tid; idx < NTOK * NTOK; idx += 384) {
      int r = idx / NTOK, c = idx - r * NTOK;
      mk[r * MKS + c] = msrc[idx];
    }
  }
  __syncthreads();

  // wave -> (which matrix, N-tile) for the QKV phase: 3 * 4 = 12 waves exactly
  const int which = wv >> 2;     // 0=Q 1=K 2=V
  const int ntw   = wv & 3;

  for (int h = 0; h < NHEAD; ++h) {
    // ---- phase A: Qh/Kh/Vh = x @ W[:, h*64 + ntw*16 .. +16] (+bias) ----
    {
      const int kdim = (which == 2) ? DIMV : CIN;
      const __bf16* wt = wst + (which == 0 ? WQT_OFF
                               : which == 1 ? WKT_OFF : WVT_OFF);
      const float* bias = (which == 0) ? bq : (which == 1) ? bk : bv;
      const int ncol = h * HDIM + ntw * 16 + l15;       // weight output channel
      const __bf16* wrow = wt + (size_t)ncol * kdim + kob;

      int r0 = l15;       r0 = (r0 < NTOK) ? r0 : NTOK;  // clamp to zero row
      int r1 = 16 + l15;  r1 = (r1 < NTOK) ? r1 : NTOK;
      int r2 = 32 + l15;  r2 = (r2 < NTOK) ? r2 : NTOK;
      int r3 = 48 + l15;  r3 = (r3 < NTOK) ? r3 : NTOK;
      const __bf16* a0 = xb + r0 * XS + koa;
      const __bf16* a1 = xb + r1 * XS + koa;
      const __bf16* a2 = xb + r2 * XS + koa;
      const __bf16* a3 = xb + r3 * XS + koa;

      v8f acc0 = {}, acc1 = {}, acc2 = {}, acc3 = {};
      for (int k0 = 0; k0 < kdim; k0 += 32) {
        v16bf bm = load_b16(wrow + k0);
        acc0 = wmma_bf16(load_a16(a0 + k0), bm, acc0);
        acc1 = wmma_bf16(load_a16(a1 + k0), bm, acc1);
        acc2 = wmma_bf16(load_a16(a2 + k0), bm, acc2);
        acc3 = wmma_bf16(load_a16(a3 + k0), bm, acc3);
      }
      const float bs = bias[ncol];
      const int nl = ntw * 16 + l15;                    // head-local channel 0..63
      if (which == 2) {
        // V transposed [d][m]: 8 accumulator rows = 8 consecutive tokens -> b128
        v8bf p0, p1, p2, p3;
#pragma unroll
        for (int r = 0; r < 8; ++r) {
          p0[r] = (__bf16)(acc0[r] + bs);
          p1[r] = (__bf16)(acc1[r] + bs);
          p2[r] = (__bf16)(acc2[r] + bs);
          p3[r] = (__bf16)(acc3[r] + bs);
        }
        __bf16* vr = vt + nl * QS + 8 * hi;
        *(v8bf*)(vr +  0) = p0;
        *(v8bf*)(vr + 16) = p1;
        *(v8bf*)(vr + 32) = p2;
        *(v8bf*)(vr + 48) = p3;
      } else {
        __bf16* dst = (which == 0) ? qh : kh;
#pragma unroll
        for (int r = 0; r < 8; ++r) {
          int mr = r + hi * 8;
          dst[(mr +  0) * QS + nl] = (__bf16)(acc0[r] + bs);
          dst[(mr + 16) * QS + nl] = (__bf16)(acc1[r] + bs);
          dst[(mr + 32) * QS + nl] = (__bf16)(acc2[r] + bs);
          dst[(mr + 48) * QS + nl] = (__bf16)(acc3[r] + bs);
        }
      }
    }
    __syncthreads();

    // ---- phase B: S = scale * Q K^T + mask (16 tiles of 16x16) ----
    for (int t = wv; t < 16; t += 12) {
      int i = t >> 2, j = t & 3;
      const __bf16* qa = qh + (16 * i + l15) * QS + koa;
      const __bf16* kb = kh + (16 * j + l15) * QS + kob;
      v8f acc = {};
      acc = wmma_bf16(load_a16(qa +  0), load_b16(kb +  0), acc);
      acc = wmma_bf16(load_a16(qa + 32), load_b16(kb + 32), acc);
#pragma unroll
      for (int r = 0; r < 8; ++r) {
        int nrow = 16 * i + r + hi * 8;
        int mcol = 16 * j + l15;
        float v = acc[r] * SCALEF;
        if (nrow < NTOK && mcol < NTOK)
          v += mk[nrow * MKS + mcol];
        else
          v = -3.0e38f;
        sb[nrow * SS + mcol] = v;
      }
    }
    __syncthreads();

    // ---- phase C: row softmax over 49 keys, emit bf16 probs ----
    if (tid < 64) {
      int row = tid;
      __bf16* prow = ph + row * QS;
      if (row < NTOK) {
        float* srow = sb + row * SS;
        float mx = -3.0e38f;
        for (int c = 0; c < NTOK; ++c) mx = fmaxf(mx, srow[c]);
        float sum = 0.f;
        for (int c = 0; c < NTOK; ++c) {
          float e = __expf(srow[c] - mx);
          srow[c] = e; sum += e;
        }
        float inv = 1.0f / sum;
        for (int c = 0; c < NTOK; ++c) prow[c] = (__bf16)(srow[c] * inv);
        for (int c = NTOK; c < 64; ++c) prow[c] = (__bf16)0.0f;
      } else {
        for (int c = 0; c < 64; ++c) prow[c] = (__bf16)0.0f;
      }
    }
    __syncthreads();

    // ---- phase D: O^T tile = V^T x P^T (A = vt rows=channels, B = ph rows=tokens)
    //      D lane layout: col = token, 8 rows = 8 consecutive channels -> b128 store
    for (int t = wv; t < 16; t += 12) {
      int i = t >> 2, dt = t & 3;          // i: token tile, dt: channel tile
      const __bf16* va = vt + (16 * dt + l15) * QS + koa;
      const __bf16* pb = ph + (16 * i + l15) * QS + kob;
      v8f acc = {};
      acc = wmma_bf16(load_a16(va +  0), load_b16(pb +  0), acc);
      acc = wmma_bf16(load_a16(va + 32), load_b16(pb + 32), acc);
      int tok = 16 * i + l15;
      v8bf pk;
#pragma unroll
      for (int r = 0; r < 8; ++r) pk[r] = (__bf16)acc[r];
      *(v8bf*)(ob + tok * OS + h * HDIM + 16 * dt + 8 * hi) = pk;
    }
    __syncthreads();
  }

  // ---- phase E: out[b] = O @ Wp + bp (4 x 64 tiles) ----
  for (int t = wv; t < 256; t += 12) {
    int i = t >> 6, n2 = t & 63;
    int ncol = n2 * 16 + l15;
    const __bf16* wrow = wst + WPT_OFF + (size_t)ncol * DIMV + kob;
    const __bf16* arow = ob + (16 * i + l15) * OS + koa;
    v8f acc = {};
    for (int k0 = 0; k0 < DIMV; k0 += 32)
      acc = wmma_bf16(load_a16(arow + k0), load_b16(wrow + k0), acc);
    float bs = bp[ncol];
#pragma unroll
    for (int r = 0; r < 8; ++r) {
      int tok = 16 * i + r + hi * 8;
      if (tok < NTOK)
        out[((size_t)b * NTOK + tok) * CIN + ncol] = acc[r] + bs;
    }
  }
}

extern "C" void kernel_launch(void* const* d_in, const int* in_sizes, int n_in,
                              void* d_out, int out_size, void* d_ws, size_t ws_size,
                              hipStream_t stream) {
  const float* x    = (const float*)d_in[0];
  const float* mask = (const float*)d_in[1];
  const float* Wq   = (const float*)d_in[2];
  const float* bq   = (const float*)d_in[3];
  const float* Wk   = (const float*)d_in[4];
  const float* bk   = (const float*)d_in[5];
  const float* Wv   = (const float*)d_in[6];
  const float* bv   = (const float*)d_in[7];
  const float* Wp   = (const float*)d_in[8];
  const float* bp   = (const float*)d_in[9];
  __bf16* ws = (__bf16*)d_ws;
  float* out = (float*)d_out;

  wattn_prep<<<(WS_HALVES + 255) / 256, 256, 0, stream>>>(Wq, Wk, Wv, Wp, ws);
  wattn_main<<<2048, 384, LDS_BYTES, stream>>>(x, mask, bq, bk, bv, bp, ws, out);
}
